// FeatureMapTransformer_81638738362845
// MI455X (gfx1250) — matrix operand order repaired and checked by reading, not compiled, not run on hardware
//
#include <hip/hip_runtime.h>
#include <hip/hip_bf16.h>

// ---------------------------------------------------------------------------
// CDNA5 (gfx1250) wave32 bf16 WMMA GEMM, double-buffered LDS pipeline with
// GLOBAL_LOAD_ASYNC_TO_LDS_B128 staging (ASYNCcnt) for bf16 K-major operands.
//
// Canonical op: D[i,j] = sum_k A[i,k] * Bop[k,j]   (A bf16, K-major [i][k])
//   BMODE 0 (NT_BF16): B is bf16 [j][k]  (K-major)  -> async global->LDS copy
//   BMODE 1 (NN_F32) : B is f32  [k][j]             -> b128 loads, pack2bf,
//                                                      packed b32 LDS stores
//   OMODE 0: f32 [i][j]           OMODE 1: bf16 [i][j]
//   OMODE 2: bf16 transposed [j][i] (packed b128 stores)
//   OMODE 3: f32 [i][j] with fused relu(gamma*v + res)
//
// Block: 256 threads = 8 wave32.  Tile BM=64 x BN=128, BK=32.
// Wave grid 2x4; each wave computes a 32x32 tile = 2x2 WMMA tiles
// -> 4 x v_wmma_f32_16x16x32_bf16 per K-step per wave, 1 barrier per K-step.
// ---------------------------------------------------------------------------

typedef __attribute__((ext_vector_type(16))) __bf16        v16bf;
typedef __attribute__((ext_vector_type(8)))  float         v8f;
typedef __attribute__((ext_vector_type(4)))  unsigned int  v4u;
typedef __attribute__((ext_vector_type(2)))  unsigned int  v2u;

union FragAB {
  v16bf v;
  v4u   u[2];
};

__device__ __forceinline__ unsigned int rne_bf(float f) {
  unsigned int u = __float_as_uint(f);
  return u + 0x7FFFu + ((u >> 16) & 1u);   // rounded; bf16 in high half
}

__device__ __forceinline__ unsigned short f2bf(float f) {
  return (unsigned short)(rne_bf(f) >> 16);
}

// pack two f32 -> {lo: bf16(a), hi: bf16(b)} in one u32
__device__ __forceinline__ unsigned int pack2bf(float a, float b) {
  return (rne_bf(a) >> 16) | (rne_bf(b) & 0xFFFF0000u);
}

// Async DMA one 16B chunk global -> LDS (tracked on ASYNCcnt).
// Generic LDS pointers carry the LDS byte offset in their low 32 bits.
__device__ __forceinline__ void async_b128(const void* gptr, const void* lptr) {
  unsigned int loff = (unsigned int)(unsigned long long)lptr;
  asm volatile("global_load_async_to_lds_b128 %0, %1, off"
               :
               : "v"(loff), "v"(gptr)
               : "memory");
}

__device__ __forceinline__ void wait_asynccnt0() {
  asm volatile("s_wait_asynccnt 0x0" ::: "memory");
}

template<int BMODE, int OMODE, bool HAS_BIAS>
__global__ __launch_bounds__(256)
void gemm_bf16_kernel(const unsigned short* __restrict__ A,   // bf16 [M][K], lda
                      const void* __restrict__ Bp,            // see BMODE
                      const float* __restrict__ bias,         // [M] or null
                      void* __restrict__ Dp,                  // see OMODE
                      int K, int lda, int ldb, int ldd,
                      const float* __restrict__ gamma,
                      const float* __restrict__ res) {
  constexpr int KS = 40;  // LDS k-stride in halves (80B, 16B aligned, pad vs banks)
  __shared__ __align__(16) unsigned short lA[2][64 * KS];
  __shared__ __align__(16) unsigned short lB[2][128 * KS];

  const int tid  = threadIdx.x;
  const int lane = tid & 31;
  const int w    = tid >> 5;   // wave 0..7
  const int wr   = w >> 2;     // 0..1 : 32-row group
  const int wc   = w & 3;      // 0..3 : 32-col group
  const int mn   = lane & 15;
  const int g    = lane >> 4;  // fragment K-half selector

  const int rowBase = blockIdx.y * 64;
  const int colBase = blockIdx.x * 128;

  // A staging: 64 rows x 32 k halves = 2048 halves = 256 threads x one b128
  const int sAr = tid >> 2;
  const int sAk = (tid & 3) * 8;

  const unsigned short* Bh = (const unsigned short*)Bp;
  const float*          Bf = (const float*)Bp;

  v8f zero = {0.f, 0.f, 0.f, 0.f, 0.f, 0.f, 0.f, 0.f};
  v8f acc[2][2] = {{zero, zero}, {zero, zero}};

  float4 rq[4];  // NN_F32 register staging: 2 patches x (2 k-rows of a j-quad)

  auto asyncStage = [&](int k0, int buf) {
    // A: one b128 DMA per thread
    async_b128(A + (size_t)(rowBase + sAr) * lda + k0 + sAk,
               &lA[buf][sAr * KS + sAk]);
    if constexpr (BMODE == 0) {
      // B bf16 [j][k]: 128 x 32 halves = 256 threads x 2 b128 DMAs
#pragma unroll
      for (int it = 0; it < 2; ++it) {
        int idx = tid + it * 256;
        int j = idx >> 2, kq = (idx & 3) * 8;
        async_b128(Bh + (size_t)(colBase + j) * ldb + k0 + kq,
                   &lB[buf][j * KS + kq]);
      }
    }
  };

  // BMODE==1: B f32 [k][j]. Each patch = k-pair x j-quad: 2 float4 loads.
  // 512 patches cover the 32x128 tile; 2 patches per thread.
  auto gloadB = [&](int k0) {
    if constexpr (BMODE == 1) {
#pragma unroll
      for (int it = 0; it < 2; ++it) {
        int idx = tid + it * 256;        // 0..511
        int jq  = (idx & 31) * 4;        // j-quad base (coalesced over lanes)
        int kp  = idx >> 5;              // k-pair 0..15
        rq[2 * it]     = *(const float4*)(Bf + (size_t)(k0 + 2 * kp) * ldb + colBase + jq);
        rq[2 * it + 1] = *(const float4*)(Bf + (size_t)(k0 + 2 * kp + 1) * ldb + colBase + jq);
      }
    }
  };
  auto lstoreB = [&](int buf) {
    if constexpr (BMODE == 1) {
#pragma unroll
      for (int it = 0; it < 2; ++it) {
        int idx = tid + it * 256;
        int jq  = (idx & 31) * 4;
        int kp  = idx >> 5;
        float4 f0 = rq[2 * it], f1 = rq[2 * it + 1];
        *(unsigned int*)&lB[buf][(jq + 0) * KS + 2 * kp] = pack2bf(f0.x, f1.x);
        *(unsigned int*)&lB[buf][(jq + 1) * KS + 2 * kp] = pack2bf(f0.y, f1.y);
        *(unsigned int*)&lB[buf][(jq + 2) * KS + 2 * kp] = pack2bf(f0.z, f1.z);
        *(unsigned int*)&lB[buf][(jq + 3) * KS + 2 * kp] = pack2bf(f0.w, f1.w);
      }
    }
  };

  // Prologue: stage tile 0
  asyncStage(0, 0);
  gloadB(0);
  lstoreB(0);
  wait_asynccnt0();
  __syncthreads();

  const int nk = K >> 5;
  for (int kt = 0; kt < nk; ++kt) {
    const int buf  = kt & 1;
    const bool more = (kt + 1 < nk);
    if (more) {
      asyncStage((kt + 1) << 5, buf ^ 1);  // DMA overlaps the WMMAs below
      gloadB((kt + 1) << 5);
    }

    // Fragment loads per CDNA5 16-bit WMMA layouts.
    // A (16x32): lane m, elems 0..7 = K 8g..8g+7, elems 8..15 = K 16+8g..+7.
    // B (32x16): lane = 16*g + n, elems 0..7 = K 8g..8g+7, elems 8..15 = K 16+8g..+7.
    FragAB af[2], bfr[2];
#pragma unroll
    for (int t = 0; t < 2; ++t) {
      const v4u* pa = (const v4u*)&lA[buf][(wr * 32 + t * 16 + mn) * KS];
      af[t].u[0] = pa[g];
      af[t].u[1] = pa[2 + g];
      const v4u* pb = (const v4u*)&lB[buf][(wc * 32 + t * 16 + mn) * KS];
      bfr[t].u[0] = pb[g];
      bfr[t].u[1] = pb[2 + g];
    }
#pragma unroll
    for (int ti = 0; ti < 2; ++ti)
#pragma unroll
      for (int tj = 0; tj < 2; ++tj)
        acc[ti][tj] = __builtin_amdgcn_wmma_f32_16x16x32_bf16(
            false, af[ti].v, false, bfr[tj].v, (short)0, acc[ti][tj], false, false);

    if (more) {
      lstoreB(buf ^ 1);
      wait_asynccnt0();
    }
    __syncthreads();
  }

  // D fragment (16x16 f32): lane l, reg r -> row = r + 8*(l>>4), col = l&15.
#pragma unroll
  for (int ti = 0; ti < 2; ++ti) {
#pragma unroll
    for (int tj = 0; tj < 2; ++tj) {
      const v8f a = acc[ti][tj];
      const int rbase = rowBase + wr * 32 + ti * 16 + 8 * g;  // rows rbase..rbase+7
      const int col   = colBase + wc * 32 + tj * 16 + mn;
      if constexpr (OMODE == 0) {
        float* D = (float*)Dp;
#pragma unroll
        for (int r = 0; r < 8; ++r) {
          float v = a[r];
          if (HAS_BIAS) v += bias[rbase + r];
          D[(size_t)(rbase + r) * ldd + col] = v;
        }
      } else if constexpr (OMODE == 3) {
        float* D = (float*)Dp;
        const float gm = gamma[0];
#pragma unroll
        for (int r = 0; r < 8; ++r) {
          float o = gm * a[r] + res[(size_t)(rbase + r) * ldd + col];
          D[(size_t)(rbase + r) * ldd + col] = o > 0.f ? o : 0.f;
        }
      } else if constexpr (OMODE == 1) {
        unsigned short* D = (unsigned short*)Dp;
#pragma unroll
        for (int r = 0; r < 8; ++r) {
          float v = a[r];
          if (HAS_BIAS) v += bias[rbase + r];
          D[(size_t)(rbase + r) * ldd + col] = f2bf(v);
        }
      } else {  // OMODE == 2: bf16 transposed [j][i], one b128 per fragment
        unsigned short* D = (unsigned short*)Dp;
        v4u pk;
#pragma unroll
        for (int h = 0; h < 4; ++h) {
          float v0 = a[2 * h], v1 = a[2 * h + 1];
          if (HAS_BIAS) {
            v0 += bias[rbase + 2 * h];
            v1 += bias[rbase + 2 * h + 1];
          }
          pk[h] = pack2bf(v0, v1);
        }
        *(v4u*)(D + (size_t)col * ldd + rbase) = pk;
      }
    }
  }
}

// One-time f32 -> bf16 pack (weights), float4 per thread.
__global__ __launch_bounds__(256)
void pack_bf16_kernel(const float* __restrict__ src, unsigned short* __restrict__ dst,
                      int n4) {
  int i = blockIdx.x * blockDim.x + threadIdx.x;
  if (i < n4) {
    float4 f = *(const float4*)(src + (size_t)i * 4);
    v2u o;
    o[0] = pack2bf(f.x, f.y);
    o[1] = pack2bf(f.z, f.w);
    *(v2u*)(dst + (size_t)i * 4) = o;
  }
}

// Row softmax in-place (f32). Logits were computed transposed so the
// reference's axis=1 softmax is over contiguous rows.
__global__ __launch_bounds__(256)
void softmax_rows_kernel(float* __restrict__ S, int ncols) {
  __shared__ float red[256];
  const int tid = threadIdx.x;
  float* p = S + (size_t)blockIdx.x * ncols;

  float mx = -3.402823e38f;
  for (int i = tid; i < ncols; i += 256) mx = fmaxf(mx, p[i]);
  red[tid] = mx;
  __syncthreads();
  for (int s = 128; s > 0; s >>= 1) {
    if (tid < s) red[tid] = fmaxf(red[tid], red[tid + s]);
    __syncthreads();
  }
  mx = red[0];
  __syncthreads();

  float sum = 0.f;
  for (int i = tid; i < ncols; i += 256) {
    float e = __expf(p[i] - mx);
    p[i] = e;
    sum += e;
  }
  red[tid] = sum;
  __syncthreads();
  for (int s = 128; s > 0; s >>= 1) {
    if (tid < s) red[tid] += red[tid + s];
    __syncthreads();
  }
  float inv = 1.f / red[0];
  for (int i = tid; i < ncols; i += 256) p[i] *= inv;
}

extern "C" void kernel_launch(void* const* d_in, const int* in_sizes, int n_in,
                              void* d_out, int out_size, void* d_ws, size_t ws_size,
                              hipStream_t stream) {
  (void)in_sizes; (void)n_in; (void)out_size; (void)ws_size;

  const float* visSegF = (const float*)d_in[0];
  const float* irSegF  = (const float*)d_in[1];
  const float* x_vis_p = (const float*)d_in[2];
  const float* x_ir_p  = (const float*)d_in[3];
  const float* Wq_rd = (const float*)d_in[4];  const float* bq_rd = (const float*)d_in[5];
  const float* Wk_rd = (const float*)d_in[6];  const float* bk_rd = (const float*)d_in[7];
  const float* Wv_rd = (const float*)d_in[8];  const float* bv_rd = (const float*)d_in[9];
  const float* Wq_dr = (const float*)d_in[10]; const float* bq_dr = (const float*)d_in[11];
  const float* Wk_dr = (const float*)d_in[12]; const float* bk_dr = (const float*)d_in[13];
  const float* Wv_dr = (const float*)d_in[14]; const float* bv_dr = (const float*)d_in[15];
  const float* gamma_rd = (const float*)d_in[16];
  const float* gamma_dr = (const float*)d_in[17];
  float* out = (float*)d_out;

  const int C = 512;
  const int N = 64 * 64;  // 4096

  // Workspace (all 16B aligned): Wbf[6] | qt | kt | vbf | S  ~= 79 MB
  unsigned short* Wbf = (unsigned short*)d_ws;                 // 6 x C*C bf16
  unsigned short* qt  = Wbf + (size_t)6 * C * C;               // [N][C] bf16
  unsigned short* kt  = qt + (size_t)N * C;                    // [N][C] bf16
  unsigned short* vbf = kt + (size_t)N * C;                    // [C][N] bf16
  float*          S   = (float*)(vbf + (size_t)C * N);         // [N][N] f32

  const float* Wsrc[6] = {Wq_rd, Wk_rd, Wv_rd, Wq_dr, Wk_dr, Wv_dr};
  for (int i = 0; i < 6; ++i) {
    pack_bf16_kernel<<<dim3((C * C / 4 + 255) / 256), dim3(256), 0, stream>>>(
        Wsrc[i], Wbf + (size_t)i * C * C, C * C / 4);
  }

  const dim3 blk(256);
  const dim3 gConv(N / 128, C / 64);  // 512 x 4096 outputs
  const dim3 gAttn(N / 128, N / 64);  // 4096 x 4096 outputs

  struct Path {
    const float *xq, *xk, *xv, *bq, *bk, *bv, *gamma, *res;
    const unsigned short *Wq, *Wk, *Wv;
    float* dst;
  };

  for (int b = 0; b < 2; ++b) {
    const size_t off = (size_t)b * C * N;
    Path paths[2] = {
      // rd: q <- x_vis_p, k <- visSegF, v <- x_ir_p, residual x_vis_p
      { x_vis_p + off, visSegF + off, x_ir_p + off,
        bq_rd, bk_rd, bv_rd, gamma_rd, x_vis_p + off,
        Wbf + (size_t)0 * C * C, Wbf + (size_t)1 * C * C, Wbf + (size_t)2 * C * C,
        out + off },
      // dr: q <- x_ir_p, k/v <- irSegF, residual x_ir_p
      { x_ir_p + off, irSegF + off, irSegF + off,
        bq_dr, bk_dr, bv_dr, gamma_dr, x_ir_p + off,
        Wbf + (size_t)3 * C * C, Wbf + (size_t)4 * C * C, Wbf + (size_t)5 * C * C,
        out + (size_t)2 * C * N + off },
    };

    for (int p = 0; p < 2; ++p) {
      const Path& P = paths[p];
      // conv q/k: A=W bf16, B=X f32 (NN), out bf16 transposed [n][c]
      gemm_bf16_kernel<1, 2, true><<<gConv, blk, 0, stream>>>(
          P.Wq, P.xq, P.bq, qt, C, C, N, C, nullptr, nullptr);
      gemm_bf16_kernel<1, 2, true><<<gConv, blk, 0, stream>>>(
          P.Wk, P.xk, P.bk, kt, C, C, N, C, nullptr, nullptr);
      // conv v: out bf16 [c][n]
      gemm_bf16_kernel<1, 1, true><<<gConv, blk, 0, stream>>>(
          P.Wv, P.xv, P.bv, vbf, C, C, N, N, nullptr, nullptr);
      // logits (transposed): S[m,n] = sum_c kt[m,c]*qt[n,c]  -- pure NT bf16
      gemm_bf16_kernel<0, 0, false><<<gAttn, blk, 0, stream>>>(
          kt, qt, nullptr, S, C, C, C, N, nullptr, nullptr);
      // softmax over query axis n == contiguous rows of S
      softmax_rows_kernel<<<N, blk, 0, stream>>>(S, N);
      // out[c,n] = sum_m vbf[c,m]*S[m,n], fused relu(gamma*out + res)
      gemm_bf16_kernel<1, 3, false><<<gConv, blk, 0, stream>>>(
          vbf, S, nullptr, P.dst, N, N, N, N, P.gamma, P.res);
    }
  }
}
